// MovingWindowHyperbolicForecaster_45552423141920
// MI455X (gfx1250) — matrix operand order repaired and checked by reading, not compiled, not run on hardware
//
#include <hip/hip_runtime.h>
#include <hip/hip_bf16.h>

typedef _Float16 v16h __attribute__((ext_vector_type(16)));
typedef _Float16 v8h  __attribute__((ext_vector_type(8)));
typedef float    v8f  __attribute__((ext_vector_type(8)));
typedef unsigned int u32x4 __attribute__((ext_vector_type(4)));
typedef int          i32x4 __attribute__((ext_vector_type(4)));
typedef int          i32x8 __attribute__((ext_vector_type(8)));

#define B_DIM 64
#define L_DIM 336
#define F_DIM 64
#define SEG   24
#define S_DIM 14
#define D_DIM 256
#define H_DIM 512
#define NROW  (B_DIM * F_DIM * 5)   // 20480 rows = BF * 5 streams
#define EPSF  1e-7f

#if __has_builtin(__builtin_amdgcn_tensor_load_to_lds) && __has_builtin(__builtin_amdgcn_s_wait_tensorcnt)
#define USE_TDM 1
#else
#define USE_TDM 0
#endif

// ---------------------------------------------------------------- helpers ---
__device__ inline float artanh_clip(float x) {
    x = fminf(x, 1.0f - 1e-6f);
    x = fmaxf(x, 0.0f);
    return 0.5f * logf((1.0f + x) / (1.0f - x));
}

// full 32-lane butterfly sum; every lane ends with the wave-wide total
__device__ inline float wave_sum(float v) {
    #pragma unroll
    for (int o = 16; o > 0; o >>= 1) v += __shfl_xor(v, o, 32);
    return v;
}

#if USE_TDM
// Issue a 2D TDM load of a 64(N) x 32(K) f16 tile of Bw[k.., n0..] into LDS.
// D# per CDNA5 ISA: group0 = count/lds_addr/global_addr/type,
// group1 = data_size + tensor dims + tile dims + dim0 stride. Groups 2/3 NULL (2D).
__device__ inline void tdm_issue_b_tile(const _Float16* Bw, int k, int n0,
                                        int N, int K, unsigned lds_addr) {
    unsigned long long ga = (unsigned long long)(const void*)(Bw + (size_t)k * N + n0);
    u32x4 g0;
    g0.x = 1u;                                                // count = 1 descriptor
    g0.y = lds_addr;                                          // LDS byte address
    g0.z = (unsigned)(ga & 0xFFFFFFFFu);                      // global_addr[31:0]
    g0.w = (unsigned)((ga >> 32) & 0x01FFFFFFu) | (2u << 30); // addr[56:32] | type=2
    unsigned td0 = (unsigned)N, td1 = (unsigned)K, s0 = (unsigned)N;
    i32x8 g1;
    g1[0] = (int)(1u << 16);                                  // data_size = 1 -> 2 bytes
    g1[1] = (int)((td0 & 0xFFFFu) << 16);                     // tensor_dim0[15:0]
    g1[2] = (int)((td0 >> 16) | ((td1 & 0xFFFFu) << 16));     // td0 hi | td1 lo
    g1[3] = (int)((td1 >> 16) | (64u << 16));                 // td1 hi | tile_dim0 = 64
    g1[4] = (int)32u;                                         // tile_dim1 = 32, tile_dim2 = 0
    g1[5] = (int)s0;                                          // tensor_dim0_stride[31:0]
    g1[6] = 0;                                                // stride0 hi | stride1 lo
    g1[7] = 0;
    i32x4 gz = {0, 0, 0, 0};
#if __clang_major__ >= 23
    i32x8 gz8 = {0, 0, 0, 0, 0, 0, 0, 0};
    __builtin_amdgcn_tensor_load_to_lds(g0, g1, gz, gz, gz8, 0);
#else
    __builtin_amdgcn_tensor_load_to_lds(g0, g1, gz, gz, 0);
#endif
}
#endif

// ------------------------------------------------------------- embedding ---
// One wave per (bf, stream, s) row: z = expmap0(tanh(xseg @ W_emb + b_emb))
__global__ __launch_bounds__(256)
void embed_kernel(const float* __restrict__ trend, const float* __restrict__ coarse,
                  const float* __restrict__ fine,  const float* __restrict__ resid,
                  const float* __restrict__ W_emb, const float* __restrict__ b_emb,
                  float* __restrict__ z) {
    int wave = threadIdx.x >> 5, lane = threadIdx.x & 31;
    int gid  = blockIdx.x * 8 + wave;          // over 4096*14*5 rows
    int stream = gid % 5;
    int rest   = gid / 5;
    int s  = rest % S_DIM;
    int bf = rest / S_DIM;
    int b  = bf >> 6, f = bf & 63;
    float xs[SEG];
    #pragma unroll
    for (int e = 0; e < SEG; e++) {            // wave-uniform loads
        int idx = (b * L_DIM + s * SEG + e) * F_DIM + f;
        float v;
        if      (stream == 0) v = trend[idx] + coarse[idx] + fine[idx] + resid[idx];
        else if (stream == 1) v = trend[idx];
        else if (stream == 2) v = coarse[idx];
        else if (stream == 3) v = fine[idx];
        else                  v = resid[idx];
        xs[e] = v;
    }
    float u[8];
    float nl = 0.f;
    #pragma unroll
    for (int i = 0; i < 8; i++) {
        int d = lane + 32 * i;
        float acc = b_emb[d];
        #pragma unroll
        for (int e = 0; e < SEG; e++) acc += xs[e] * W_emb[e * D_DIM + d];
        u[i] = tanhf(acc);
        nl += u[i] * u[i];
    }
    float vn = sqrtf(fmaxf(wave_sum(nl), EPSF * EPSF));
    float t  = tanhf(vn);                      // c == 1
    float scale = t / vn;                      // |z| == t
    float maxn = 1.0f - 1e-5f;
    if (t > maxn) scale *= maxn / t;           // project()
    int row = bf * 5 + stream;
    float* zo = z + ((size_t)row * S_DIM + s) * D_DIM;
    #pragma unroll
    for (int i = 0; i < 8; i++) zo[lane + 32 * i] = u[i] * scale;
}

// ----------------------------------------------- velocity + logmap0(last) ---
// One wave per row; slot pairs streamed (cur/nxt rotate) -> ~50 VGPRs, 0 barriers
__global__ __launch_bounds__(256)
void velocity_kernel(const float* __restrict__ z, _Float16* __restrict__ A1, int it) {
    int wave = threadIdx.x >> 5, lane = threadIdx.x & 31;
    int row = blockIdx.x * 8 + wave;
    const float* zr = z + (size_t)row * S_DIM * D_DIM;
    float cur[8], nxt[8], vel[8], w[8];
    float nl = 0.f;
    #pragma unroll
    for (int i = 0; i < 8; i++) {
        vel[i] = 0.f;
        cur[i] = zr[(size_t)(it % S_DIM) * D_DIM + lane + 32 * i];
        nl += cur[i] * cur[i];
    }
    float x2 = wave_sum(nl);
    // decay weights: w[t] = 0.9^(12-t) / sum_{i=0..12} 0.9^i
    float Z = 0.f, p = 1.f;
    #pragma unroll
    for (int i = 0; i < S_DIM - 1; i++) { Z += p; p *= 0.9f; }
    float pw = 1.f;
    #pragma unroll
    for (int i = 0; i < S_DIM - 2; i++) pw *= 0.9f;   // 0.9^12
    #pragma unroll
    for (int t = 0; t < S_DIM - 1; t++) {
        int phys = (t + 1 + it) % S_DIM;
        float y2l = 0.f, xyl = 0.f;
        #pragma unroll
        for (int i = 0; i < 8; i++) {
            nxt[i] = zr[(size_t)phys * D_DIM + lane + 32 * i];
            y2l += nxt[i] * nxt[i];
            xyl += cur[i] * nxt[i];
        }
        float y2 = wave_sum(y2l);
        float xy = wave_sum(xyl);
        // mobius_add(-x, y): dot(-x,y) = -xy
        float den = fmaxf(1.f - 2.f * xy + x2 * y2, 1e-15f);
        float ca  = 1.f - 2.f * xy + y2;
        float cb  = 1.f - x2;
        float wnl = 0.f;
        #pragma unroll
        for (int i = 0; i < 8; i++) {
            w[i] = (ca * (-cur[i]) + cb * nxt[i]) / den;
            wnl += w[i] * w[i];
        }
        float wn   = sqrtf(fmaxf(wave_sum(wnl), EPSF * EPSF));
        float lam  = 2.f / fmaxf(1.f - x2, EPSF);
        float coef = (2.f / lam) * artanh_clip(wn) / wn;
        float wt   = pw / Z;
        #pragma unroll
        for (int i = 0; i < 8; i++) { vel[i] += wt * coef * w[i]; cur[i] = nxt[i]; }
        pw /= 0.9f;
        x2 = y2;
    }
    // logmap0(z_last); cur now holds the last slot, x2 its squared norm
    float yn = sqrtf(fmaxf(x2, EPSF * EPSF));
    float lc = artanh_clip(yn) / yn;
    _Float16* a = A1 + (size_t)row * (2 * D_DIM);
    #pragma unroll
    for (int i = 0; i < 8; i++) {
        a[lane + 32 * i]         = (_Float16)(lc * cur[i]);
        a[D_DIM + lane + 32 * i] = (_Float16)vel[i];
    }
}

// ------------------------------------- expmap(z_last, m) + logmap0(z_next) ---
__global__ __launch_bounds__(256)
void expmap_kernel(float* __restrict__ z, const float* __restrict__ m,
                   _Float16* __restrict__ A3, int it) {
    int wave = threadIdx.x >> 5, lane = threadIdx.x & 31;
    int row = blockIdx.x * 8 + wave;
    float* zr = z + (size_t)row * S_DIM * D_DIM;
    const float* mr = m + (size_t)row * D_DIM;
    float x[8], v[8], sec[8], zn[8];
    float x2l = 0.f, v2l = 0.f;
    int pl = (S_DIM - 1 + it) % S_DIM;
    #pragma unroll
    for (int i = 0; i < 8; i++) {
        x[i] = zr[(size_t)pl * D_DIM + lane + 32 * i];
        v[i] = mr[lane + 32 * i];
        x2l += x[i] * x[i];
        v2l += v[i] * v[i];
    }
    float x2 = wave_sum(x2l);
    float vn = sqrtf(fmaxf(wave_sum(v2l), EPSF * EPSF));
    float lam = 2.f / fmaxf(1.f - x2, EPSF);
    float th  = tanhf(lam * vn * 0.5f);
    float sf  = th / vn;
    float y2l = 0.f, xyl = 0.f;
    #pragma unroll
    for (int i = 0; i < 8; i++) {
        sec[i] = sf * v[i];
        y2l += sec[i] * sec[i];
        xyl += x[i] * sec[i];
    }
    float y2 = wave_sum(y2l);
    float xy = wave_sum(xyl);
    float den = fmaxf(1.f + 2.f * xy + x2 * y2, 1e-15f);
    float ca  = 1.f + 2.f * xy + y2;
    float cb  = 1.f - x2;
    float nl = 0.f;
    #pragma unroll
    for (int i = 0; i < 8; i++) {
        zn[i] = (ca * x[i] + cb * sec[i]) / den;
        nl += zn[i] * zn[i];
    }
    float n = sqrtf(fmaxf(wave_sum(nl), EPSF * EPSF));
    float maxn = 1.0f - 1e-5f;
    float sc = (n > maxn) ? (maxn / n) : 1.0f;
    float nn = fminf(n, maxn);
    float lc = artanh_clip(nn) / nn;
    float*    zo = zr + (size_t)(it % S_DIM) * D_DIM;   // overwrite vacated slot
    _Float16* ao = A3 + (size_t)row * D_DIM;
    #pragma unroll
    for (int i = 0; i < 8; i++) {
        float zp = zn[i] * sc;
        zo[lane + 32 * i] = zp;
        ao[lane + 32 * i] = (_Float16)(lc * zp);        // logmap0(z_next)
    }
}

// ----------------------------------------------------------- WMMA GEMM -----
// C(MxN) = A(MxK,f16) * B(KxN,f16) + bias. 8 wave32s, 128x64 block tile.
// Double-buffered TDM pipeline: tile s+1 streams into LDS buf (s+1)&1 while
// all 8 waves run WMMA on tile s in buf s&1. TDM completes in order, so
// s_wait_tensorcnt(1) after issuing the next tile guarantees tile s landed.
enum { ACT_TANH = 0, ACT_F32 = 1, ACT_RELU = 2, ACT_OUT = 3 };

template<int ACT>
__global__ __launch_bounds__(256)
void gemm_wmma(const _Float16* __restrict__ A, const _Float16* __restrict__ Bw,
               const float* __restrict__ bias, void* __restrict__ outp,
               int N, int K, int it) {
    __shared__ __align__(16) _Float16 Bs[2][32 * 64];   // 2 x 4 KB
    int n0   = blockIdx.x * 64;
    int m0   = blockIdx.y * 128;
    int tid  = threadIdx.x;
    int lane = tid & 31;
    int wave = tid >> 5;
    // A-fragment addressing per ISA 16-bit A layout: lane%16 = M, lane/16 = K-half
    int mrow = m0 + wave * 16 + (lane & 15);
    int kh   = lane >> 4;
    const int nsteps = K >> 5;                 // K/32; always even here
    v8f acc[4] = {};
#if USE_TDM
    unsigned lds0 = (unsigned)(unsigned long long)(void*)&Bs[0][0];
    unsigned lds1 = lds0 + 32 * 64 * 2;
    if (tid == 0) tdm_issue_b_tile(Bw, 0, n0, N, K, lds0);   // prologue: tile 0
#endif
    auto stage = [&](int s, int buf) {
        __syncthreads();                       // other buffer's readers are done
#if USE_TDM
        if (tid == 0) {
            if (s + 1 < nsteps) {
                tdm_issue_b_tile(Bw, (s + 1) << 5, n0, N, K, buf ? lds0 : lds1);
                __builtin_amdgcn_s_wait_tensorcnt(1);   // tile s done; s+1 in flight
            } else {
                __builtin_amdgcn_s_wait_tensorcnt(0);   // drain final tile
            }
        }
#else
        {   // cooperative fallback: 256 threads x 16B into Bs[buf]
            int kk = tid >> 3;
            int ng = (tid & 7) * 8;
            *(v8h*)(&Bs[buf][0] + kk * 64 + ng) =
                *(const v8h*)(Bw + (size_t)((s << 5) + kk) * N + n0 + ng);
        }
#endif
        __syncthreads();                       // tile s visible to all 8 waves
        int k = s << 5;
        const _Float16* ar = A + (size_t)mrow * K + k + kh * 8;
        v8h a0 = *(const v8h*)(ar);
        v8h a1 = *(const v8h*)(ar + 16);
        if (k + 32 < K) __builtin_prefetch(ar + 32, 0, 3);   // next A tile, near cache
        v16h afrag;
        #pragma unroll
        for (int i = 0; i < 8; i++) { afrag[i] = a0[i]; afrag[8 + i] = a1[i]; }
        // fetch all four B fragments first, then issue WMMAs back-to-back
        v16h bfrag[4];
        #pragma unroll
        for (int j = 0; j < 4; j++) {
            const _Float16* br = &Bs[buf][0] + (kh * 8) * 64 + j * 16 + (lane & 15); // lane%16 = N, lane/16 = K-half
            #pragma unroll
            for (int i = 0; i < 8; i++) { bfrag[j][i] = br[i * 64]; bfrag[j][8 + i] = br[(16 + i) * 64]; }
        }
        #pragma unroll
        for (int j = 0; j < 4; j++)
            acc[j] = __builtin_amdgcn_wmma_f32_16x16x32_f16(
                false, afrag, false, bfrag[j], (short)0, acc[j], false, false);
    };
    for (int s = 0; s < nsteps; s += 2) {      // unrolled pipeline, static parity
        stage(s, 0);
        stage(s + 1, 1);
    }
    // epilogue per ISA f32 C/D layout: lane%16 = N, lane/16 -> M+8, vgpr r = M row
    int mb  = m0 + wave * 16 + (lane >> 4) * 8;
    int nit = lane & 15;
    #pragma unroll
    for (int j = 0; j < 4; j++) {
        int ncol = n0 + j * 16 + nit;
        float bsv = bias[ncol];
        #pragma unroll
        for (int r = 0; r < 8; r++) {
            int mr = mb + r;
            float val = acc[j][r] + bsv;
            if constexpr (ACT == ACT_TANH) {
                ((_Float16*)outp)[(size_t)mr * N + ncol] = (_Float16)tanhf(val);
            } else if constexpr (ACT == ACT_F32) {
                ((float*)outp)[(size_t)mr * N + ncol] = val;
            } else if constexpr (ACT == ACT_RELU) {
                ((_Float16*)outp)[(size_t)mr * N + ncol] = (_Float16)fmaxf(val, 0.f);
            } else {  // ACT_OUT: scatter into (5, B, 96, F); cols 24..63 are padding
                if (ncol < SEG) {
                    int bf = mr / 5, stream = mr % 5;
                    int b = bf >> 6, f = bf & 63;
                    ((float*)outp)[(((size_t)stream * B_DIM + b) * 96 +
                                    (it * SEG + ncol)) * F_DIM + f] = val;
                }
            }
        }
    }
}

// ------------------------------------------------------- weight prep -------
__global__ void cvt_f16(const float* __restrict__ s, _Float16* __restrict__ d, int n) {
    int i = blockIdx.x * 256 + threadIdx.x;
    if (i < n) d[i] = (_Float16)s[i];
}
__global__ void pad_wr2(const float* __restrict__ w, const float* __restrict__ bias,
                        _Float16* __restrict__ wp, float* __restrict__ bp) {
    int i = blockIdx.x * 256 + threadIdx.x;   // over 512*64
    if (i < 512 * 64) {
        int kk = i >> 6, nn = i & 63;
        wp[i] = (nn < SEG) ? (_Float16)w[kk * SEG + nn] : (_Float16)0.f;
        if (i < 64) bp[i] = (i < SEG) ? bias[i] : 0.f;
    }
}

// --------------------------------------------------------------- launch ----
extern "C" void kernel_launch(void* const* d_in, const int* in_sizes, int n_in,
                              void* d_out, int out_size, void* d_ws, size_t ws_size,
                              hipStream_t stream) {
    const float* trend  = (const float*)d_in[0];
    const float* coarse = (const float*)d_in[1];
    const float* fine   = (const float*)d_in[2];
    const float* resid  = (const float*)d_in[3];
    const float* W_emb  = (const float*)d_in[4];
    const float* b_emb  = (const float*)d_in[5];
    const float* W1     = (const float*)d_in[6];
    const float* b1     = (const float*)d_in[7];
    const float* W2     = (const float*)d_in[8];
    const float* b2     = (const float*)d_in[9];
    const float* Wr1    = (const float*)d_in[10];
    const float* br1    = (const float*)d_in[11];
    const float* Wr2    = (const float*)d_in[12];
    const float* br2    = (const float*)d_in[13];
    (void)in_sizes; (void)n_in; (void)out_size; (void)ws_size;

    char* ws = (char*)d_ws;
    size_t off = 0;
    auto take = [&](size_t bytes) -> char* {
        char* p = ws + off;
        off = (off + bytes + 255) & ~(size_t)255;
        return p;
    };
    float*    z    = (float*)   take((size_t)NROW * S_DIM * D_DIM * 4);  // 293.6 MB
    _Float16* A1   = (_Float16*)take((size_t)NROW * 2 * D_DIM * 2);      // 21 MB
    _Float16* h    = (_Float16*)take((size_t)NROW * H_DIM * 2);          // 21 MB
    float*    mbuf = (float*)   take((size_t)NROW * D_DIM * 4);          // 21 MB
    _Float16* A3   = (_Float16*)take((size_t)NROW * D_DIM * 2);          // 10.5 MB
    _Float16* r1   = (_Float16*)take((size_t)NROW * H_DIM * 2);          // 21 MB
    _Float16* W1h  = (_Float16*)take(512 * 512 * 2);
    _Float16* W2h  = (_Float16*)take(512 * 256 * 2);
    _Float16* Wr1h = (_Float16*)take(256 * 512 * 2);
    _Float16* Wr2h = (_Float16*)take(512 * 64 * 2);
    float*    br2p = (float*)   take(64 * 4);

    cvt_f16<<<(512 * 512 + 255) / 256, 256, 0, stream>>>(W1,  W1h,  512 * 512);
    cvt_f16<<<(512 * 256 + 255) / 256, 256, 0, stream>>>(W2,  W2h,  512 * 256);
    cvt_f16<<<(256 * 512 + 255) / 256, 256, 0, stream>>>(Wr1, Wr1h, 256 * 512);
    pad_wr2<<<(512 * 64 + 255) / 256, 256, 0, stream>>>(Wr2, br2, Wr2h, br2p);

    embed_kernel<<<(B_DIM * F_DIM * S_DIM * 5) / 8, 256, 0, stream>>>(
        trend, coarse, fine, resid, W_emb, b_emb, z);

    for (int it = 0; it < 4; it++) {
        velocity_kernel<<<NROW / 8, 256, 0, stream>>>(z, A1, it);
        gemm_wmma<ACT_TANH><<<dim3(512 / 64, NROW / 128), 256, 0, stream>>>(
            A1, W1h, b1, h, 512, 512, it);
        gemm_wmma<ACT_F32><<<dim3(256 / 64, NROW / 128), 256, 0, stream>>>(
            h, W2h, b2, mbuf, 256, 512, it);
        expmap_kernel<<<NROW / 8, 256, 0, stream>>>(z, mbuf, A3, it);
        gemm_wmma<ACT_RELU><<<dim3(512 / 64, NROW / 128), 256, 0, stream>>>(
            A3, Wr1h, br1, r1, 512, 256, it);
        gemm_wmma<ACT_OUT><<<dim3(1, NROW / 128), 256, 0, stream>>>(
            r1, Wr2h, br2p, d_out, 64, 512, it);
    }
}
